// GCNWH_12232066859465
// MI455X (gfx1250) — matrix operand order repaired and checked by reading, not compiled, not run on hardware
//
#include <hip/hip_runtime.h>
#include <hip/hip_bf16.h>
#include <stdint.h>

// ---------------------------------------------------------------------------
// CDNA5 WMMA types (gfx1250, wave32)
// ---------------------------------------------------------------------------
typedef __attribute__((ext_vector_type(16))) __bf16 v16bf;
typedef __attribute__((ext_vector_type(8)))  float  v8f;

union V16BF {
    v16bf     v;
    uint32_t  d[8];
};

__device__ __forceinline__ uint32_t bf16_rne_bits(float f) {
    union { float f; uint32_t u; } x;
    x.f = f;
    uint32_t r = x.u + 0x7FFFu + ((x.u >> 16) & 1u);   // round-to-nearest-even
    return r >> 16;
}

__device__ __forceinline__ uint32_t pack2_bf16(float lo, float hi) {
    return bf16_rne_bits(lo) | (bf16_rne_bits(hi) << 16);
}

// ---------------------------------------------------------------------------
// Degree accumulation + rsqrt
// ---------------------------------------------------------------------------
__global__ void gcn_deg_kernel(const long long* __restrict__ col,
                               const float* __restrict__ ew,
                               float* __restrict__ deg, int E) {
    int i = blockIdx.x * blockDim.x + threadIdx.x;
    if (i < E) atomicAdd(&deg[col[i]], ew[i]);
}

__global__ void gcn_dis_kernel(float* __restrict__ deg, int N) {
    int i = blockIdx.x * blockDim.x + threadIdx.x;
    if (i < N) deg[i] = rsqrtf(deg[i] + 1.0f);
}

// ---------------------------------------------------------------------------
// Pre-pack: fp32 features -> packed bf16 pairs (row-major, K/2 dwords per row)
// ---------------------------------------------------------------------------
__global__ void gcn_packX_kernel(const float2* __restrict__ X2,
                                 uint32_t* __restrict__ Xp, int ndw) {
    int t = blockIdx.x * blockDim.x + threadIdx.x;
    if (t >= ndw) return;
    float2 ab = X2[t];
    Xp[t] = pack2_bf16(ab.x, ab.y);
}

// ---------------------------------------------------------------------------
// Pre-pack: W[K x Ncols] fp32 -> B fragments in WMMA lane layout.
// Bpack[((ct*(K/32)+k32)*32 + lane)*8 + j] packs K = k32*32 + 16*half + 2j
// (lo) and K+1 (hi) for column ct*16 + (lane&15).
// ---------------------------------------------------------------------------
__global__ void gcn_packW_kernel(const float* __restrict__ W,
                                 uint32_t* __restrict__ Bpack,
                                 int K, int Ncols) {
    int t = blockIdx.x * blockDim.x + threadIdx.x;
    int total = (Ncols / 16) * (K / 32) * 32 * 8;
    if (t >= total) return;
    int j    = t & 7;
    int lane = (t >> 3) & 31;
    int rest = t >> 8;
    int k32  = rest % (K / 32);
    int ct   = rest / (K / 32);
    int half = lane >> 4;
    int n    = lane & 15;
    int k    = k32 * 32 + 16 * half + 2 * j;
    float s0 = W[(size_t)k * Ncols + ct * 16 + n];
    float s1 = W[(size_t)(k + 1) * Ncols + ct * 16 + n];
    Bpack[t] = pack2_bf16(s0, s1);
}

// ---------------------------------------------------------------------------
// WMMA GEMM: Out[Nrows x NCOLS] = Apack(bf16) * Bpack(bf16), f32 accumulate.
// One wave per 16-row strip; strip reuses its A fragment across all NCOLS/16
// column tiles.  A: two b128 loads per K-step; B: two b128 loads per tile
// from the pre-swizzled (L2-resident) pack.  All strides compile-time.
// ---------------------------------------------------------------------------
template <int NCOLS, int K>
__global__ void gcn_gemm_wmma(const uint32_t* __restrict__ Apack,  // [Nrows, K/2]
                              const uint32_t* __restrict__ Bpack,
                              float* __restrict__ Out, int Nrows) {
    constexpr int KK32 = K / 32;
    constexpr int CT   = NCOLS / 16;

    const int wave = threadIdx.x >> 5;
    const int lane = threadIdx.x & 31;
    const int rt   = blockIdx.x * (blockDim.x >> 5) + wave;   // 16-row strip id
    const int row0 = rt * 16;
    if (row0 >= Nrows) return;

    const int mn   = lane & 15;
    const int half = lane >> 4;

    int arow = row0 + mn;
    if (arow >= Nrows) arow = Nrows - 1;                      // clamp tail loads
    const uint32_t* __restrict__ abase = Apack + (size_t)arow * (K / 2);
    const uint32_t* __restrict__ bbase = Bpack + (size_t)lane * 8;

    v8f acc[CT];
#pragma unroll
    for (int c = 0; c < CT; ++c) acc[c] = v8f{};

#pragma unroll
    for (int kk32 = 0; kk32 < KK32; ++kk32) {
        V16BF a;
        {   // A fragment: dwords (K pairs) 8*half .. 8*half+7 (K = 16*half + s)
            const uint4* p0 = reinterpret_cast<const uint4*>(abase + kk32 * 16 + 8 * half);
            const uint4* p1 = reinterpret_cast<const uint4*>(abase + kk32 * 16 + 8 * half + 4);
            uint4 A0 = *p0, A1 = *p1;
            a.d[0] = A0.x; a.d[1] = A0.y; a.d[2] = A0.z; a.d[3] = A0.w;
            a.d[4] = A1.x; a.d[5] = A1.y; a.d[6] = A1.z; a.d[7] = A1.w;
        }
#pragma unroll
        for (int ct = 0; ct < CT; ++ct) {
            V16BF b;
            const uint32_t* bp = bbase + (size_t)(ct * KK32 + kk32) * 256;
            const uint4* q0 = reinterpret_cast<const uint4*>(bp);
            const uint4* q1 = reinterpret_cast<const uint4*>(bp + 4);
            uint4 B0 = *q0, B1 = *q1;
            b.d[0] = B0.x; b.d[1] = B0.y; b.d[2] = B0.z; b.d[3] = B0.w;
            b.d[4] = B1.x; b.d[5] = B1.y; b.d[6] = B1.z; b.d[7] = B1.w;
            acc[ct] = __builtin_amdgcn_wmma_f32_16x16x32_bf16(
                false, a.v, false, b.v, (short)0, acc[ct], false, false);
        }
    }

    // C/D layout: VGPR j -> row (j + 8*half), col = lane&15
    if (row0 + 16 <= Nrows) {
        float* __restrict__ o = Out + (size_t)(row0 + 8 * half) * NCOLS + mn;
#pragma unroll
        for (int ct = 0; ct < CT; ++ct)
#pragma unroll
            for (int j = 0; j < 8; ++j)
                o[(size_t)j * NCOLS + ct * 16] = acc[ct][j];
    } else {
#pragma unroll
        for (int ct = 0; ct < CT; ++ct)
#pragma unroll
            for (int j = 0; j < 8; ++j) {
                int r = row0 + j + 8 * half;
                if (r < Nrows) Out[(size_t)r * NCOLS + ct * 16 + mn] = acc[ct][j];
            }
    }
}

// ---------------------------------------------------------------------------
// Edge scatter: one wave per edge; lane owns D/32 contiguous features
// (b128/b64 gather), atomics resolve in L2 (dst fits in 192MB L2).
// ---------------------------------------------------------------------------
template <int D>
__global__ void gcn_agg_kernel(const long long* __restrict__ row,
                               const long long* __restrict__ col,
                               const float* __restrict__ ew,
                               const float* __restrict__ dis,
                               const float* __restrict__ src,
                               float* __restrict__ dst, int E) {
    int gid  = blockIdx.x * blockDim.x + threadIdx.x;
    int e    = gid >> 5;
    int lane = gid & 31;
    if (e >= E) return;
    long long r = row[e];
    long long c = col[e];
    float nrm = dis[r] * ew[e] * dis[c];
    constexpr int V = D / 32;
    const float* __restrict__ s = src + (size_t)r * D + lane * V;
    float* __restrict__ d = dst + (size_t)c * D + lane * V;
    if constexpr (V == 4) {
        float4 v = *reinterpret_cast<const float4*>(s);
        atomicAdd(&d[0], nrm * v.x);
        atomicAdd(&d[1], nrm * v.y);
        atomicAdd(&d[2], nrm * v.z);
        atomicAdd(&d[3], nrm * v.w);
    } else {
        float2 v = *reinterpret_cast<const float2*>(s);
        atomicAdd(&d[0], nrm * v.x);
        atomicAdd(&d[1], nrm * v.y);
    }
}

// ---------------------------------------------------------------------------
// finalize layer1: h = relu(agg + dis^2*xw + b1), emitted as packed bf16
// ---------------------------------------------------------------------------
__global__ void gcn_fin1_kernel(const float* __restrict__ agg,
                                const float* __restrict__ xw,
                                const float* __restrict__ dis,
                                const float* __restrict__ b,
                                uint32_t* __restrict__ hpack,
                                int N, int D) {
    int t = blockIdx.x * blockDim.x + threadIdx.x;
    int total = N * (D / 2);
    if (t >= total) return;
    int node = t / (D / 2);
    int f0   = (t % (D / 2)) * 2;
    float ds = dis[node];
    float d2 = ds * ds;
    size_t base = (size_t)node * D + f0;
    float v0 = fmaxf(agg[base]     + d2 * xw[base]     + b[f0],     0.0f);
    float v1 = fmaxf(agg[base + 1] + d2 * xw[base + 1] + b[f0 + 1], 0.0f);
    hpack[t] = pack2_bf16(v0, v1);
}

// finalize layer2: out += dis^2 * hw + b2
__global__ void gcn_fin2_kernel(float* __restrict__ out,
                                const float* __restrict__ hw,
                                const float* __restrict__ dis,
                                const float* __restrict__ b,
                                int N, int D) {
    int i = blockIdx.x * blockDim.x + threadIdx.x;
    if (i >= N * D) return;
    int node = i / D;
    int f    = i % D;
    float ds = dis[node];
    out[i] = out[i] + ds * ds * hw[i] + b[f];
}

// ---------------------------------------------------------------------------
// Launcher
// ---------------------------------------------------------------------------
extern "C" void kernel_launch(void* const* d_in, const int* in_sizes, int n_in,
                              void* d_out, int out_size, void* d_ws, size_t ws_size,
                              hipStream_t stream) {
    const float*     x   = (const float*)d_in[0];
    const long long* ei  = (const long long*)d_in[1];   // int64 [2, E]
    const float*     ew  = (const float*)d_in[2];
    const float*     W1  = (const float*)d_in[3];
    const float*     b1  = (const float*)d_in[4];
    const float*     W2  = (const float*)d_in[5];
    const float*     b2  = (const float*)d_in[6];
    float*           out = (float*)d_out;

    const int DIN = 128, DH = 128, DOUT = 64;
    const int N = in_sizes[0] / DIN;
    const int E = in_sizes[1] / 2;

    const long long* row = ei;
    const long long* col = ei + E;

    // ---- workspace layout (dwords) ----
    // dis | pack (x-bf16 then h-bf16) | xw/hw f32 | agg f32 | W1pack | W2pack
    uint32_t* ws = (uint32_t*)d_ws;
    size_t o_dis  = 0;
    size_t o_pack = (((size_t)N + 511) & ~(size_t)511);          // N dwords
    size_t o_xw   = o_pack + (size_t)N * (DIN / 2);               // N*64 dwords
    size_t o_agg  = o_xw   + (size_t)N * DH;                      // N*128 dwords
    size_t o_w1p  = o_agg  + (size_t)N * DH;                      // N*128 dwords
    size_t o_w2p  = o_w1p  + (size_t)(DH / 16) * (DIN / 32) * 256;

    float*    dis   = (float*)(ws + o_dis);
    uint32_t* pack  = ws + o_pack;     // x-bf16 for layer1, h-bf16 for layer2
    float*    xw    = (float*)(ws + o_xw);   // xw for layer1, hw for layer2
    float*    agg   = (float*)(ws + o_agg);
    uint32_t* w1p   = ws + o_w1p;
    uint32_t* w2p   = ws + o_w2p;

    hipMemsetAsync(dis, 0, (size_t)N * sizeof(float), stream);
    hipMemsetAsync(agg, 0, (size_t)N * DH * sizeof(float), stream);
    hipMemsetAsync(out, 0, (size_t)N * DOUT * sizeof(float), stream);

    const int TB = 256;

    // gcn_norm
    gcn_deg_kernel<<<(E + TB - 1) / TB, TB, 0, stream>>>(col, ew, dis, E);
    gcn_dis_kernel<<<(N + TB - 1) / TB, TB, 0, stream>>>(dis, N);

    // pre-pack x and weights into bf16 fragment form
    {
        int ndw = N * (DIN / 2);
        gcn_packX_kernel<<<(ndw + TB - 1) / TB, TB, 0, stream>>>(
            (const float2*)x, pack, ndw);
        int w1dw = (DH / 16) * (DIN / 32) * 256;
        gcn_packW_kernel<<<(w1dw + TB - 1) / TB, TB, 0, stream>>>(W1, w1p, DIN, DH);
        int w2dw = (DOUT / 16) * (DH / 32) * 256;
        gcn_packW_kernel<<<(w2dw + TB - 1) / TB, TB, 0, stream>>>(W2, w2p, DH, DOUT);
    }

    const int WPB    = 4;                       // waves (row strips) per block
    const int strips = (N + 15) / 16;
    const int gblk   = (strips + WPB - 1) / WPB;

    // ---- layer 1 ----
    gcn_gemm_wmma<128, 128><<<gblk, 32 * WPB, 0, stream>>>(pack, w1p, xw, N);
    {
        long long threads = (long long)E * 32;
        gcn_agg_kernel<128><<<(int)((threads + TB - 1) / TB), TB, 0, stream>>>(
            row, col, ew, dis, xw, agg, E);
    }
    {
        int total = N * (DH / 2);
        gcn_fin1_kernel<<<(total + TB - 1) / TB, TB, 0, stream>>>(
            agg, xw, dis, b1, pack, N, DH);     // pack now holds h (bf16)
    }

    // ---- layer 2 ----  (hw reuses the xw buffer; scatter straight into out)
    gcn_gemm_wmma<64, 128><<<gblk, 32 * WPB, 0, stream>>>(pack, w2p, xw, N);
    {
        long long threads = (long long)E * 32;
        gcn_agg_kernel<64><<<(int)((threads + TB - 1) / TB), TB, 0, stream>>>(
            row, col, ew, dis, xw, out, E);
    }
    gcn_fin2_kernel<<<((long long)N * DOUT + TB - 1) / TB, TB, 0, stream>>>(
        out, xw, dis, b2, N, DOUT);
}
